// SchNetPlus_11089605559055
// MI455X (gfx1250) — compile-verified, matrix operand order checked by reading, not run on hardware
//
#include <hip/hip_runtime.h>
#include <hip/hip_bf16.h>

typedef __bf16 bf16_t;
typedef __attribute__((ext_vector_type(16))) __bf16 v16bf;
typedef __attribute__((ext_vector_type(8)))  __bf16 v8bf;
typedef __attribute__((ext_vector_type(8)))  float   v8f;

#define HID 128
#define NG  50
#define NGP 64
#define WST 136   // padded K-stride for LDS-resident weights / staging
#define SLOPE 0.015f

__device__ __forceinline__ float lrelu(float x) { return x > 0.f ? x : SLOPE * x; }

// Load a 16x32 bf16 A (or B^T) fragment from a row-major [16 x stride] buffer.
// ISA 16-bit A layout: lane<16 -> row=lane, K=[0..7]+[16..23]; lane>=16 -> row=lane-16, K=[8..15]+[24..31].
__device__ __forceinline__ v16bf ld_frag(const bf16_t* p, int stride, int lane) {
  const int r  = lane & 15;
  const int kh = (lane >> 4) & 1;
  const bf16_t* q = p + r * stride + kh * 8;
  v8bf lo = *(const v8bf*)(q);
  v8bf hi = *(const v8bf*)(q + 16);
  v16bf out;
#pragma unroll
  for (int i = 0; i < 8; ++i) { out[i] = lo[i]; out[8 + i] = hi[i]; }
  return out;
}

__device__ __forceinline__ v8f wmma_bf16(v16bf a, v16bf b, v8f c) {
  return __builtin_amdgcn_wmma_f32_16x16x32_bf16(false, a, false, b, (short)0, c, false, false);
}

// Build the gaussian-smearing A fragment directly in registers:
// element i of the fragment corresponds to K = kb*32 + kh*8 + i + (i>=8 ? 8 : 0).
__device__ __forceinline__ v16bf demb_frag(float d, bool ok, int kb, int kh) {
  const float step  = 5.0f / 49.0f;
  const float coeff = -0.5f / (step * step);
  v16bf a;
#pragma unroll
  for (int i = 0; i < 16; ++i) {
    int k = kb * 32 + kh * 8 + i + ((i >= 8) ? 8 : 0);
    float val = 0.f;
    if (k < NG && ok) {
      float t = d - (float)k * step;
      val = __expf(coeff * t * t);
    }
    a[i] = (bf16_t)val;
  }
  return a;
}

// ---------------- small prep kernels ----------------

__global__ void k_edge_geom(const int* __restrict__ ei, const float* __restrict__ pos,
                            float* __restrict__ dist, float* __restrict__ Cc, int E) {
  int e = blockIdx.x * blockDim.x + threadIdx.x;
  if (e >= E) return;
  int r = ei[e], c = ei[E + e];
  float dx = pos[3*r+0] - pos[3*c+0];
  float dy = pos[3*r+1] - pos[3*c+1];
  float dz = pos[3*r+2] - pos[3*c+2];
  float d = sqrtf(dx*dx + dy*dy + dz*dz);
  dist[e] = d;
  Cc[e]   = 0.5f * (cosf(d * 0.6283185307179586f) + 1.0f);  // pi/5
}

__global__ void k_embed(const int* __restrict__ z, const float* __restrict__ emb,
                        float* __restrict__ v, bf16_t* __restrict__ vbf, int N) {
  int i = blockIdx.x * blockDim.x + threadIdx.x;
  if (i >= N * HID) return;
  int a = i >> 7, c = i & (HID - 1);
  float x = emb[z[a] * HID + c];
  v[i] = x;
  vbf[i] = (bf16_t)x;
}

// Transpose + bf16-pack weights: W[K][Nn] (row major) -> Wt[Nn][KP], zero pad K..KP.
__global__ void k_pack(const float* __restrict__ W, bf16_t* __restrict__ Wt,
                       int K, int KP, int Nn) {
  int i = blockIdx.x * blockDim.x + threadIdx.x;
  if (i >= Nn * KP) return;
  int n = i / KP, k = i - n * KP;
  float x = (k < K) ? W[k * Nn + n] : 0.f;
  Wt[i] = (bf16_t)x;
}

// ---------------- node linear: T = v @ w_lin (bf16 out) ----------------

__global__ void __launch_bounds__(128) k_node_lin(const bf16_t* __restrict__ Abf,
                                                  const bf16_t* __restrict__ Wt,
                                                  bf16_t* __restrict__ Tout, int N) {
  const int lane = threadIdx.x & 31;
  const int wv   = threadIdx.x >> 5;
  const int row0 = (blockIdx.x * 4 + wv) * 16;
  if (row0 >= N) return;
  const int nIdx = lane & 15;
  const int m0   = (lane >> 4) * 8;
  for (int ct = 0; ct < 8; ++ct) {
    v8f acc = {0,0,0,0,0,0,0,0};
#pragma unroll
    for (int kb = 0; kb < 4; ++kb) {
      v16bf a = ld_frag(Abf + row0 * HID + kb * 32, HID, lane);
      v16bf b = ld_frag(Wt + (ct * 16) * HID + kb * 32, HID, lane);
      acc = wmma_bf16(a, b, acc);
    }
    int n = ct * 16 + nIdx;
#pragma unroll
    for (int vv = 0; vv < 8; ++vv)
      Tout[(row0 + m0 + vv) * HID + n] = (bf16_t)acc[vv];
  }
}

// ---------------- fused per-edge layer (persistent, weights in LDS) ----------------
// h1 = lrelu(demb@w1+b1); h2 = lrelu(h1@w2+b2); W = (h2@w3+b3)*C;
// e = T[row]*W; e2 = e@lout + lb + e; atomicAdd(agg[col], e2)

__global__ void __launch_bounds__(128) k_edge(const int* __restrict__ ei,
        const float* __restrict__ dist, const float* __restrict__ Cc,
        const bf16_t* __restrict__ Tbf,
        const bf16_t* __restrict__ w1t, const bf16_t* __restrict__ gw2,
        const bf16_t* __restrict__ gw3, const bf16_t* __restrict__ glo,
        const float* __restrict__ b1, const float* __restrict__ b2,
        const float* __restrict__ b3, const float* __restrict__ lb,
        float* __restrict__ agg, int E) {
  __shared__ __align__(16) bf16_t sW2[HID * WST];
  __shared__ __align__(16) bf16_t sW3[HID * WST];
  __shared__ __align__(16) bf16_t sLO[HID * WST];
  __shared__ __align__(16) bf16_t sS0[4][16][WST];
  __shared__ __align__(16) bf16_t sS1[4][16][WST];
  __shared__ __align__(16) bf16_t sT [4][16][HID];
  __shared__ int    sRow[4][16];
  __shared__ int    sCol[4][16];
  __shared__ float  sC  [4][16];

  // cooperative block-wide weight staging (bf16 already packed/transposed)
  {
    const int nvec = HID * WST * (int)sizeof(bf16_t) / 16;  // uint4 per matrix
    uint4* d2 = (uint4*)sW2; const uint4* s2 = (const uint4*)gw2;
    uint4* d3 = (uint4*)sW3; const uint4* s3 = (const uint4*)gw3;
    uint4* dl = (uint4*)sLO; const uint4* sl = (const uint4*)glo;
    for (int i = threadIdx.x; i < nvec; i += 128) {
      d2[i] = s2[i];
      d3[i] = s3[i];
      dl[i] = sl[i];
    }
  }
  __syncthreads();   // only barrier; everything below is per-wave

  const int lane = threadIdx.x & 31;
  const int wv   = threadIdx.x >> 5;
  const int nIdx = lane & 15;
  const int m0   = (lane >> 4) * 8;
  const int kh   = (lane >> 4) & 1;
  const int tilesTotal = (E + 15) / 16;

  bf16_t* s0 = &sS0[wv][0][0];
  bf16_t* s1 = &sS1[wv][0][0];

  for (int tile = blockIdx.x * 4 + wv; tile < tilesTotal; tile += gridDim.x * 4) {
    const int e0 = tile * 16;

    // edge metadata (lanes 0..15); same-wave LDS ops are in order on CDNA5
    if (lane < 16) {
      int e = e0 + lane;
      bool ok = e < E;
      sRow[wv][lane] = ok ? ei[e]     : -1;
      sCol[wv][lane] = ok ? ei[E + e] : -1;
      sC[wv][lane]   = ok ? Cc[e] : 0.f;
    }

    // stage T[row] rows (16 x 128 bf16)
#pragma unroll 4
    for (int j = 0; j < 64; ++j) {
      int idx = j * 32 + lane;
      int er = idx >> 7, c = idx & 127;
      int r = sRow[wv][er];
      sT[wv][er][c] = (r >= 0) ? Tbf[r * HID + c] : (bf16_t)0.f;
    }

    // per-lane distance for the register-built gaussian A fragment
    const int   eRow = e0 + (lane & 15);
    const bool  okRow = eRow < E;
    const float dRow  = okRow ? dist[eRow] : 0.f;

    // MM1: h1 = lrelu(demb @ w1 + b1)  (K = 64, A built in registers)
    for (int ct = 0; ct < 8; ++ct) {
      v8f acc = {0,0,0,0,0,0,0,0};
#pragma unroll
      for (int kb = 0; kb < 2; ++kb) {
        v16bf a = demb_frag(dRow, okRow, kb, kh);
        v16bf b = ld_frag(w1t + (ct * 16) * NGP + kb * 32, NGP, lane);
        acc = wmma_bf16(a, b, acc);
      }
      int n = ct * 16 + nIdx;
      float bias = b1[n];
#pragma unroll
      for (int vv = 0; vv < 8; ++vv)
        s0[(m0 + vv) * WST + n] = (bf16_t)lrelu(acc[vv] + bias);
    }

    // MM2: h2 = lrelu(h1 @ w2 + b2)  (K = 128, B from LDS)
    for (int ct = 0; ct < 8; ++ct) {
      v8f acc = {0,0,0,0,0,0,0,0};
#pragma unroll
      for (int kb = 0; kb < 4; ++kb) {
        v16bf a = ld_frag(s0 + kb * 32, WST, lane);
        v16bf b = ld_frag(sW2 + (ct * 16) * WST + kb * 32, WST, lane);
        acc = wmma_bf16(a, b, acc);
      }
      int n = ct * 16 + nIdx;
      float bias = b2[n];
#pragma unroll
      for (int vv = 0; vv < 8; ++vv)
        s1[(m0 + vv) * WST + n] = (bf16_t)lrelu(acc[vv] + bias);
    }

    // MM3: W = (h2 @ w3 + b3) * C ; e = T[row] * W  -> s0 (reuse)
    for (int ct = 0; ct < 8; ++ct) {
      v8f acc = {0,0,0,0,0,0,0,0};
#pragma unroll
      for (int kb = 0; kb < 4; ++kb) {
        v16bf a = ld_frag(s1 + kb * 32, WST, lane);
        v16bf b = ld_frag(sW3 + (ct * 16) * WST + kb * 32, WST, lane);
        acc = wmma_bf16(a, b, acc);
      }
      int n = ct * 16 + nIdx;
      float bias = b3[n];
#pragma unroll
      for (int vv = 0; vv < 8; ++vv) {
        int M = m0 + vv;
        float x = (acc[vv] + bias) * sC[wv][M];
        x *= (float)sT[wv][M][n];
        s0[M * WST + n] = (bf16_t)x;
      }
    }

    // MM4: e2 = e @ lout + lb + e ; scatter-add to agg[col]
    for (int ct = 0; ct < 8; ++ct) {
      v8f acc = {0,0,0,0,0,0,0,0};
#pragma unroll
      for (int kb = 0; kb < 4; ++kb) {
        v16bf a = ld_frag(s0 + kb * 32, WST, lane);
        v16bf b = ld_frag(sLO + (ct * 16) * WST + kb * 32, WST, lane);
        acc = wmma_bf16(a, b, acc);
      }
      int n = ct * 16 + nIdx;
      float bias = lb[n];
#pragma unroll
      for (int vv = 0; vv < 8; ++vv) {
        int M = m0 + vv;
        int cI = sCol[wv][M];
        if (cI >= 0) {
          float x = acc[vv] + bias + (float)s0[M * WST + n];
          atomicAdd(&agg[cI * HID + n], x);
        }
      }
    }
  }
}

// ---------------- node update: v += lrelu(agg@vw1+b1)@vw2 + b2 ----------------

__global__ void __launch_bounds__(128) k_node_update(const float* __restrict__ agg,
        const bf16_t* __restrict__ vw1t, const bf16_t* __restrict__ vw2t,
        const float* __restrict__ vb1, const float* __restrict__ vb2,
        float* __restrict__ v, bf16_t* __restrict__ vbf, int N) {
  __shared__ __align__(16) bf16_t sA[4][16][WST];
  __shared__ __align__(16) bf16_t sH[4][16][WST];
  const int lane = threadIdx.x & 31;
  const int wv   = threadIdx.x >> 5;
  const int row0 = (blockIdx.x * 4 + wv) * 16;
  if (row0 >= N) return;
  const int nIdx = lane & 15;
  const int m0   = (lane >> 4) * 8;

#pragma unroll 4
  for (int j = 0; j < 64; ++j) {
    int idx = j * 32 + lane;
    int r = idx >> 7, c = idx & 127;
    sA[wv][r][c] = (bf16_t)agg[(row0 + r) * HID + c];
  }

  bf16_t* a0 = &sA[wv][0][0];
  bf16_t* h0 = &sH[wv][0][0];
  for (int ct = 0; ct < 8; ++ct) {
    v8f acc = {0,0,0,0,0,0,0,0};
#pragma unroll
    for (int kb = 0; kb < 4; ++kb) {
      v16bf a = ld_frag(a0 + kb * 32, WST, lane);
      v16bf b = ld_frag(vw1t + (ct * 16) * HID + kb * 32, HID, lane);
      acc = wmma_bf16(a, b, acc);
    }
    int n = ct * 16 + nIdx;
    float bias = vb1[n];
#pragma unroll
    for (int vv = 0; vv < 8; ++vv)
      h0[(m0 + vv) * WST + n] = (bf16_t)lrelu(acc[vv] + bias);
  }

  for (int ct = 0; ct < 8; ++ct) {
    v8f acc = {0,0,0,0,0,0,0,0};
#pragma unroll
    for (int kb = 0; kb < 4; ++kb) {
      v16bf a = ld_frag(h0 + kb * 32, WST, lane);
      v16bf b = ld_frag(vw2t + (ct * 16) * HID + kb * 32, HID, lane);
      acc = wmma_bf16(a, b, acc);
    }
    int n = ct * 16 + nIdx;
    float bias = vb2[n];
#pragma unroll
    for (int vv = 0; vv < 8; ++vv) {
      int o = (row0 + m0 + vv) * HID + n;
      float x = v[o] + acc[vv] + bias;
      v[o] = x;
      vbf[o] = (bf16_t)x;
    }
  }
}

// ---------------- readout: per-graph mean of lrelu(v@uw1+ub1)@uw2+ub2 ----------------

__global__ void __launch_bounds__(128) k_readout(const float* __restrict__ v,
        const float* __restrict__ uw1, const float* __restrict__ ub1,
        const float* __restrict__ uw2, const float* __restrict__ ub2,
        float* __restrict__ out, int per) {
  __shared__ float sW1[HID * 64];
  __shared__ float sred[128];
  int tid = threadIdx.x;
  for (int i = tid; i < HID * 64; i += blockDim.x) sW1[i] = uw1[i];
  __syncthreads();

  float s_local = 0.f;
  for (int a = tid; a < per; a += blockDim.x) {
    int atom = blockIdx.x * per + a;
    float vr[HID];
#pragma unroll
    for (int c = 0; c < HID; ++c) vr[c] = v[atom * HID + c];
    float s = ub2[0];
#pragma unroll 1
    for (int j = 0; j < 64; ++j) {
      float d = ub1[j];
#pragma unroll
      for (int c = 0; c < HID; ++c) d += vr[c] * sW1[c * 64 + j];
      s += lrelu(d) * uw2[j];
    }
    s_local += s;
  }
  sred[tid] = s_local;
  __syncthreads();
  for (int off = 64; off > 0; off >>= 1) {
    if (tid < off) sred[tid] += sred[tid + off];
    __syncthreads();
  }
  if (tid == 0) out[blockIdx.x] = sred[0] / (float)(per > 0 ? per : 1);
}

// ---------------- host ----------------

static inline size_t align_up(size_t x, size_t a) { return (x + a - 1) & ~(a - 1); }

extern "C" void kernel_launch(void* const* d_in, const int* in_sizes, int n_in,
                              void* d_out, int out_size, void* d_ws, size_t ws_size,
                              hipStream_t stream) {
  const int*   z      = (const int*)d_in[0];
  const float* pos    = (const float*)d_in[1];
  const int*   ei     = (const int*)d_in[3];
  const float* emb    = (const float*)d_in[4];
  const float* w_lin  = (const float*)d_in[5];
  const float* mw1    = (const float*)d_in[6];
  const float* mb1    = (const float*)d_in[7];
  const float* mw2    = (const float*)d_in[8];
  const float* mb2    = (const float*)d_in[9];
  const float* mw3    = (const float*)d_in[10];
  const float* mb3    = (const float*)d_in[11];
  const float* lout_w = (const float*)d_in[12];
  const float* lout_b = (const float*)d_in[13];
  const float* vw1    = (const float*)d_in[14];
  const float* vb1    = (const float*)d_in[15];
  const float* vw2    = (const float*)d_in[16];
  const float* vb2    = (const float*)d_in[17];
  const float* uw1    = (const float*)d_in[18];
  const float* ub1    = (const float*)d_in[19];
  const float* uw2    = (const float*)d_in[20];
  const float* ub2    = (const float*)d_in[21];
  (void)n_in; (void)ws_size;

  const int N = in_sizes[0];
  const int E = in_sizes[3] / 2;
  const int L = in_sizes[5] / (HID * HID);

  char* wsb = (char*)d_ws;
  size_t off = 0;
  auto carve = [&](size_t bytes) -> void* {
    void* p = wsb + off;
    off = align_up(off + bytes, 256);
    return p;
  };
  float*  v_f   = (float*)carve((size_t)N * HID * 4);
  bf16_t* vbf   = (bf16_t*)carve((size_t)N * HID * 2);
  bf16_t* tbf   = (bf16_t*)carve((size_t)N * HID * 2);
  float*  agg   = (float*)carve((size_t)N * HID * 4);
  float*  dist  = (float*)carve((size_t)E * 4);
  float*  Cc    = (float*)carve((size_t)E * 4);
  bf16_t* wlinT = (bf16_t*)carve((size_t)L * HID * HID * 2);
  bf16_t* w1T   = (bf16_t*)carve((size_t)L * HID * NGP * 2);
  bf16_t* w2T   = (bf16_t*)carve((size_t)L * HID * WST * 2);
  bf16_t* w3T   = (bf16_t*)carve((size_t)L * HID * WST * 2);
  bf16_t* loT   = (bf16_t*)carve((size_t)L * HID * WST * 2);
  bf16_t* vw1T  = (bf16_t*)carve((size_t)L * HID * HID * 2);
  bf16_t* vw2T  = (bf16_t*)carve((size_t)L * HID * HID * 2);

  const int thr = 256;
  if (E > 0)
    k_edge_geom<<<(E + thr - 1) / thr, thr, 0, stream>>>(ei, pos, dist, Cc, E);
  k_embed<<<((size_t)N * HID + thr - 1) / thr, thr, 0, stream>>>(z, emb, v_f, vbf, N);

  for (int l = 0; l < L; ++l) {
    k_pack<<<(HID * HID + thr - 1) / thr, thr, 0, stream>>>(w_lin + (size_t)l * HID * HID, wlinT + (size_t)l * HID * HID, HID, HID, HID);
    k_pack<<<(HID * NGP + thr - 1) / thr, thr, 0, stream>>>(mw1 + (size_t)l * NG * HID, w1T + (size_t)l * HID * NGP, NG, NGP, HID);
    k_pack<<<(HID * WST + thr - 1) / thr, thr, 0, stream>>>(mw2 + (size_t)l * HID * HID, w2T + (size_t)l * HID * WST, HID, WST, HID);
    k_pack<<<(HID * WST + thr - 1) / thr, thr, 0, stream>>>(mw3 + (size_t)l * HID * HID, w3T + (size_t)l * HID * WST, HID, WST, HID);
    k_pack<<<(HID * WST + thr - 1) / thr, thr, 0, stream>>>(lout_w + (size_t)l * HID * HID, loT + (size_t)l * HID * WST, HID, WST, HID);
    k_pack<<<(HID * HID + thr - 1) / thr, thr, 0, stream>>>(vw1 + (size_t)l * HID * HID, vw1T + (size_t)l * HID * HID, HID, HID, HID);
    k_pack<<<(HID * HID + thr - 1) / thr, thr, 0, stream>>>(vw2 + (size_t)l * HID * HID, vw2T + (size_t)l * HID * HID, HID, HID, HID);
  }

  const int nodeTiles  = (N + 15) / 16;
  const int nodeBlocks = (nodeTiles + 3) / 4;
  const int edgeTiles  = (E + 15) / 16;
  int edgeBlocks = (edgeTiles + 3) / 4;
  if (edgeBlocks > 2048) edgeBlocks = 2048;   // persistent grid; weights staged once per block
  if (edgeBlocks < 1) edgeBlocks = 1;

  for (int l = 0; l < L; ++l) {
    k_node_lin<<<nodeBlocks, 128, 0, stream>>>(vbf, wlinT + (size_t)l * HID * HID, tbf, N);
    hipMemsetAsync(agg, 0, (size_t)N * HID * sizeof(float), stream);
    if (E > 0)
      k_edge<<<edgeBlocks, 128, 0, stream>>>(ei, dist, Cc, tbf,
          w1T + (size_t)l * HID * NGP, w2T + (size_t)l * HID * WST,
          w3T + (size_t)l * HID * WST, loT + (size_t)l * HID * WST,
          mb1 + (size_t)l * HID, mb2 + (size_t)l * HID,
          mb3 + (size_t)l * HID, lout_b + (size_t)l * HID,
          agg, E);
    k_node_update<<<nodeBlocks, 128, 0, stream>>>(agg,
        vw1T + (size_t)l * HID * HID, vw2T + (size_t)l * HID * HID,
        vb1 + (size_t)l * HID, vb2 + (size_t)l * HID, v_f, vbf, N);
  }

  const int per = N / (out_size > 0 ? out_size : 1);
  k_readout<<<out_size, 128, 0, stream>>>(v_f, uw1, ub1, uw2, ub2, (float*)d_out, per);
}
